// NumericalMarkowitz_43422119362990
// MI455X (gfx1250) — matrix operand order repaired and checked by reading, not compiled, not run on hardware
//
#include <hip/hip_runtime.h>
#include <math.h>

// ---------------------------------------------------------------------------
// NumericalMarkowitz on MI455X (gfx1250):
//   - prep kernel: sig_sym = sigma + sigma^T, stored fp16 (134 MB, fits L2)
//   - solve kernel: 256 persistent workgroups (1 per batch), each runs the
//     full 100-step Adam loop; matvec via v_wmma_f32_16x16x32_f16; all Adam
//     state in LDS; deterministic tree reductions (no atomics).
// ---------------------------------------------------------------------------

typedef _Float16 v16h  __attribute__((ext_vector_type(16)));
typedef _Float16 half8 __attribute__((ext_vector_type(8)));
typedef float    v8f   __attribute__((ext_vector_type(8)));

#define B_DIM   256
#define N_DIM   512
#define NSTEP   100

// A-matrix fragment for V_WMMA_F32_16X16X32_F16 (16x32, row-major source):
// lane L (hi = L>>4) holds row (L&15), K = 8*hi + [0..7] in VGPRs 0-3 and
// K = 16 + 8*hi + [0..7] in VGPRs 4-7  -> two contiguous 16B segments.
__device__ __forceinline__ v16h ld_a_frag(const _Float16* p) {
    half8 s0 = *(const half8*)p;          // K = 8*hi   + 0..7
    half8 s1 = *(const half8*)(p + 16);   // K = 16+8*hi + 0..7
    return __builtin_shufflevector(s0, s1, 0,1,2,3,4,5,6,7,8,9,10,11,12,13,14,15);
}

// ---------------------------------------------------------------------------
// Prep: Sh[b][i][j] = fp16( sigma[b][i][j] + sigma[b][j][i] )
// 32x32 tiles with an LDS transpose buffer so both reads are coalesced.
// grid = (16, 16, 256), block = 256
// ---------------------------------------------------------------------------
__global__ __launch_bounds__(256)
void prep_sym(const float* __restrict__ sigma, _Float16* __restrict__ Sh) {
    __shared__ float T[32][33];
    const int tid = threadIdx.x;
    const int b   = blockIdx.z;
    const int i0  = blockIdx.y << 5;
    const int j0  = blockIdx.x << 5;
    const float*   Sg = sigma + ((size_t)b << 18);   // b * 512 * 512
    _Float16*      So = Sh    + ((size_t)b << 18);

    const int c  = tid & 31;
    const int r0 = tid >> 5;           // 8 rows per pass, 4 passes
#pragma unroll
    for (int p = 0; p < 4; ++p) {
        const int r = r0 + (p << 3);
        T[r][c] = Sg[(size_t)(j0 + r) * N_DIM + i0 + c];   // transposed tile
    }
    __syncthreads();
#pragma unroll
    for (int p = 0; p < 4; ++p) {
        const int r = r0 + (p << 3);
        const float v = Sg[(size_t)(i0 + r) * N_DIM + j0 + c] + T[c][r];
        So[(size_t)(i0 + r) * N_DIM + j0 + c] = (_Float16)v;
    }
}

// ---------------------------------------------------------------------------
// Persistent solver: one workgroup per batch, 256 threads (8 wave32),
// each wave owns 64 rows of the matvec (4 WMMA accumulators).
// ---------------------------------------------------------------------------
__global__ __launch_bounds__(256)
void markowitz_solve(const _Float16* __restrict__ Sh,
                     const float* __restrict__ mu,
                     float* __restrict__ out) {
    __shared__ __align__(32) _Float16 wh[N_DIM];  // fp16 w for WMMA B-operand
    __shared__ float wf[N_DIM];                   // fp32 w
    __shared__ float ml[N_DIM];                   // Adam m
    __shared__ float vl[N_DIM];                   // Adam v
    __shared__ float yl[N_DIM];                   // y = S w
    __shared__ float mul[N_DIM];                  // mu slice
    __shared__ float red[256];                    // reduction buffer

    const int tid  = threadIdx.x;
    const int b    = blockIdx.x;
    const int lane = tid & 31;
    const int wv   = tid >> 5;        // wave id 0..7
    const int hi   = lane >> 4;       // 0 or 1
    const int mrow = lane & 15;       // row within 16-row tile / D column id
    const int rowBase = wv << 6;      // 64 rows per wave

    const _Float16* Sb = Sh + ((size_t)b << 18);
    const int i0 = tid, i1 = tid + 256;

    // --- init state ---
    const float w0 = 1.0f / (float)N_DIM;
    wf[i0] = w0;  wf[i1] = w0;
    wh[i0] = (_Float16)w0;  wh[i1] = (_Float16)w0;
    ml[i0] = 0.f; ml[i1] = 0.f;
    vl[i0] = 0.f; vl[i1] = 0.f;
    mul[i0] = mu[(size_t)b * N_DIM + i0];
    mul[i1] = mu[(size_t)b * N_DIM + i1];
    __syncthreads();

    float p1 = 1.f, p2 = 1.f;
    for (int t = 0; t < NSTEP; ++t) {
        p1 *= 0.9f;  p2 *= 0.999f;
        const float bc1 = 1.f - p1;
        const float bc2 = 1.f - p2;

        // --- sum(w): deterministic tree reduction ---
        red[tid] = wf[i0] + wf[i1];
        __syncthreads();
        for (int s = 128; s > 0; s >>= 1) {
            if (tid < s) red[tid] += red[tid + s];
            __syncthreads();
        }
        const float dsum = red[0] - 1.0f;
        const float gs = dsum > 0.f ? (100.f / 256.f)
                       : (dsum < 0.f ? -(100.f / 256.f) : 0.f);

        // --- y = sig_sym @ w via WMMA, 64 rows per wave ---
        v8f acc0 = {}, acc1 = {}, acc2 = {}, acc3 = {};
#pragma unroll 4
        for (int kt = 0; kt < 16; ++kt) {
            const int k0 = kt << 5;
            // B operand: w[k0..k0+31] replicated across the 16 N columns:
            // lane holds K = k0 + 8*hi + [0..7] and k0 + 16 + 8*hi + [0..7]
            const v16h bf = ld_a_frag(&wh[k0 + (hi << 3)]);
            const _Float16* ra = Sb + (size_t)(rowBase + mrow) * N_DIM
                                    + k0 + (hi << 3);
            const v16h a0 = ld_a_frag(ra);
            const v16h a1 = ld_a_frag(ra + 16 * N_DIM);
            const v16h a2 = ld_a_frag(ra + 32 * N_DIM);
            const v16h a3 = ld_a_frag(ra + 48 * N_DIM);
            acc0 = __builtin_amdgcn_wmma_f32_16x16x32_f16(false, a0, false, bf,
                                                          (short)0, acc0, false, false);
            acc1 = __builtin_amdgcn_wmma_f32_16x16x32_f16(false, a1, false, bf,
                                                          (short)0, acc1, false, false);
            acc2 = __builtin_amdgcn_wmma_f32_16x16x32_f16(false, a2, false, bf,
                                                          (short)0, acc2, false, false);
            acc3 = __builtin_amdgcn_wmma_f32_16x16x32_f16(false, a3, false, bf,
                                                          (short)0, acc3, false, false);
        }
        // D layout: VGPR r, lanes 0-15 -> M=r, lanes 16-31 -> M=8+r; N = lane&15.
        // Column N==0 (mrow==0) carries the matvec result.
        if (mrow == 0) {
            const int base = rowBase + (hi << 3);
#pragma unroll
            for (int r = 0; r < 8; ++r) {
                yl[base +      r] = acc0[r];
                yl[base + 16 + r] = acc1[r];
                yl[base + 32 + r] = acc2[r];
                yl[base + 48 + r] = acc3[r];
            }
        }
        __syncthreads();

        // --- Adam update + clamp; accumulate clamped sum ---
        float s2 = 0.f;
#pragma unroll
        for (int e = 0; e < 2; ++e) {
            const int i = tid + (e << 8);
            const float w = wf[i];
            const float g = (yl[i] - mul[i]) * (1.0f / 256.0f) + gs
                          + (100.0f / (256.0f * 512.0f))
                            * ((w > 0.1f ? 1.f : 0.f) - (w < 0.f ? 1.f : 0.f));
            const float m = 0.9f   * ml[i] + 0.1f   * g;
            const float v = 0.999f * vl[i] + 0.001f * g * g;
            ml[i] = m;  vl[i] = v;
            float wn = w - 0.01f * (m / bc1) / (sqrtf(v / bc2) + 1e-8f);
            wn = fminf(fmaxf(wn, 0.f), 0.1f);
            wf[i] = wn;
            s2 += wn;
        }
        __syncthreads();

        // --- sum of clamped w, then renormalize ---
        red[tid] = s2;
        __syncthreads();
        for (int s = 128; s > 0; s >>= 1) {
            if (tid < s) red[tid] += red[tid + s];
            __syncthreads();
        }
        const float inv = 1.0f / red[0];
#pragma unroll
        for (int e = 0; e < 2; ++e) {
            const int i = tid + (e << 8);
            const float wn = wf[i] * inv;
            wf[i] = wn;
            wh[i] = (_Float16)wn;
        }
        __syncthreads();
    }

    out[(size_t)b * N_DIM + i0] = wf[i0];
    out[(size_t)b * N_DIM + i1] = wf[i1];
}

// ---------------------------------------------------------------------------
// Launch: prep (one pass over HBM) + single persistent solve kernel.
// Workspace: 256*512*512 fp16 = 134,217,728 bytes for sig_sym.
// ---------------------------------------------------------------------------
extern "C" void kernel_launch(void* const* d_in, const int* in_sizes, int n_in,
                              void* d_out, int out_size, void* d_ws, size_t ws_size,
                              hipStream_t stream) {
    (void)in_sizes; (void)n_in; (void)out_size; (void)ws_size;
    const float* mu    = (const float*)d_in[0];   // (256, 512)   f32
    const float* sigma = (const float*)d_in[1];   // (256,512,512) f32
    float*       out   = (float*)d_out;           // (256, 512)   f32
    _Float16*    Sh    = (_Float16*)d_ws;         // fp16 sig_sym, 134 MB

    prep_sym<<<dim3(16, 16, 256), 256, 0, stream>>>(sigma, Sh);
    markowitz_solve<<<dim3(256), 256, 0, stream>>>(Sh, mu, out);
}
